// Llama4TextMoe_11020886082289
// MI455X (gfx1250) — matrix operand order repaired and checked by reading, not compiled, not run on hardware
//
#include <hip/hip_runtime.h>
#include <hip/hip_bf16.h>

// ---------------- problem constants (B=2,S=512 -> T=1024) ----------------
constexpr int T_TOK = 1024;   // tokens
constexpr int HDIM  = 2048;   // hidden
constexpr int IDIM  = 4096;   // intermediate
constexpr int NEXP  = 8;      // experts

// ---------------- WMMA types ----------------
typedef __attribute__((ext_vector_type(16))) _Float16 v16h;
typedef __attribute__((ext_vector_type(8)))  float    v8f;

// GEMM tiling: 256 threads = 8 wave32, each wave owns 16 rows x 64 cols.
// TK=64: one LDS stage feeds two WMMA k-steps; double-buffered stages.
constexpr int BM = 128;
constexpr int BN = 64;
constexpr int TK = 64;
constexpr int ASZ = BM * TK;   // halves per A stage
constexpr int BSZ = BN * TK;   // halves per B stage

__device__ __forceinline__ v8f wmma16(v16h a, v16h b, v8f c) {
  // D = A(16x32 f16) x B(32x16 f16) + C(16x16 f32)
  return __builtin_amdgcn_wmma_f32_16x16x32_f16(
      /*neg_a=*/false, a, /*neg_b=*/false, b,
      /*c_mod=*/(short)0, c, /*reuse_a=*/false, /*reuse_b=*/false);
}

// ---- gfx1250 async global->LDS copy (ASYNCcnt-tracked, bypasses VGPRs) ----
__device__ __forceinline__ void async_copy_b128(const _Float16* gsrc, _Float16* ldst) {
  const unsigned lds = (unsigned)(unsigned long long)(uintptr_t)ldst; // low 32b = LDS offset
  const unsigned long long ga = (unsigned long long)(uintptr_t)gsrc;
  asm volatile("global_load_async_to_lds_b128 %0, %1, off"
               :: "v"(lds), "v"(ga) : "memory");
}
__device__ __forceinline__ void wait_async0() {
  asm volatile("s_wait_asynccnt 0" ::: "memory");
}

// A fragment: 16x32 f16, lane L holds row m0+(L&15); lanes<16 get K {0..7,16..23},
// lanes>=16 get K {8..15,24..31}  (CDNA5 16-bit A layout). ks = 0/1 -> k-slice.
__device__ __forceinline__ v16h ld_frag_a(const _Float16* lA, int m0, int ks, int lane) {
  const int m  = m0 + (lane & 15);
  const int kb = ks * 32 + ((lane & 16) ? 8 : 0);
  const _Float16* p = lA + m * TK + kb;
  union { uint4 u[2]; v16h v; } t;
  t.u[0] = *(const uint4*)(p);
  t.u[1] = *(const uint4*)(p + 16);
  return t.v;
}

// B fragment: 32x16 f16, LDS tile N-major ([BN][TK]); lane L holds column
// n0+(L&15); lanes<16 get K {0..7,16..23}, lanes>=16 get K {8..15,24..31}
__device__ __forceinline__ v16h ld_frag_b(const _Float16* lB, int n0, int ks, int lane) {
  const int n  = n0 + (lane & 15);
  const int kb = ks * 32 + ((lane & 16) ? 8 : 0);
  const _Float16* p = lB + n * TK + kb;
  union { uint4 u[2]; v16h v; } t;
  t.u[0] = *(const uint4*)(p);
  t.u[1] = *(const uint4*)(p + 16);
  return t.v;
}

// convert+store one B register set (16 f32) into an N-major LDS tile column strip
__device__ __forceinline__ void store_b_tile(_Float16* dst /* + bc*TK + bk */,
                                             const float4& f0, const float4& f1,
                                             const float4& f2, const float4& f3) {
  dst[ 0*TK]=(_Float16)f0.x; dst[ 1*TK]=(_Float16)f0.y; dst[ 2*TK]=(_Float16)f0.z; dst[ 3*TK]=(_Float16)f0.w;
  dst[ 4*TK]=(_Float16)f1.x; dst[ 5*TK]=(_Float16)f1.y; dst[ 6*TK]=(_Float16)f1.z; dst[ 7*TK]=(_Float16)f1.w;
  dst[ 8*TK]=(_Float16)f2.x; dst[ 9*TK]=(_Float16)f2.y; dst[10*TK]=(_Float16)f2.z; dst[11*TK]=(_Float16)f2.w;
  dst[12*TK]=(_Float16)f3.x; dst[13*TK]=(_Float16)f3.y; dst[14*TK]=(_Float16)f3.z; dst[15*TK]=(_Float16)f3.w;
}

// ---------------- small kernels ----------------
__global__ void moe_init_kernel(int* counts, int* fill) {
  int i = threadIdx.x;
  if (i < NEXP) { counts[i] = 0; fill[i] = 0; }
}

// one block per token: logits = x[t] @ router_w, top-1, sigmoid -> router_scores[E,T]
__global__ __launch_bounds__(256) void router_kernel(
    const float* __restrict__ x, const float* __restrict__ rw,
    float* __restrict__ scores_out, int* __restrict__ expert_of,
    float* __restrict__ score, int* __restrict__ counts) {
  const int t = blockIdx.x;
  __shared__ float red[256 * NEXP];
  float acc[NEXP];
#pragma unroll
  for (int e = 0; e < NEXP; ++e) acc[e] = 0.f;
  const float* xr = x + (size_t)t * HDIM;
  for (int h = threadIdx.x; h < HDIM; h += 256) {
    const float xv = xr[h];
    const float* w = rw + (size_t)h * NEXP;
#pragma unroll
    for (int e = 0; e < NEXP; ++e) acc[e] += xv * w[e];
  }
#pragma unroll
  for (int e = 0; e < NEXP; ++e) red[threadIdx.x * NEXP + e] = acc[e];
  __syncthreads();
  for (int s = 128; s > 0; s >>= 1) {
    if (threadIdx.x < (unsigned)s) {
#pragma unroll
      for (int e = 0; e < NEXP; ++e)
        red[threadIdx.x * NEXP + e] += red[(threadIdx.x + s) * NEXP + e];
    }
    __syncthreads();
  }
  __shared__ int   bestS;
  __shared__ float sigS;
  if (threadIdx.x == 0) {
    float best = red[0]; int bi = 0;
#pragma unroll
    for (int e = 1; e < NEXP; ++e)
      if (red[e] > best) { best = red[e]; bi = e; }   // strict > == first-max (jax tie rule)
    const float sig = 1.f / (1.f + __expf(-best));
    expert_of[t] = bi; score[t] = sig;
    atomicAdd(&counts[bi], 1);
    bestS = bi; sigS = sig;
  }
  __syncthreads();
  if (threadIdx.x < NEXP)
    scores_out[(size_t)threadIdx.x * T_TOK + t] = (threadIdx.x == (unsigned)bestS) ? sigS : 0.f;
}

__global__ void offsets_kernel(const int* counts, int* offs) {
  if (threadIdx.x == 0) {
    int s = 0;
    for (int e = 0; e < NEXP; ++e) { offs[e] = s; s += counts[e]; }
  }
}

__global__ void scatter_kernel(const int* __restrict__ expert_of,
                               const int* __restrict__ offs,
                               int* __restrict__ fill, int* __restrict__ perm) {
  const int t = blockIdx.x * 256 + threadIdx.x;
  if (t < T_TOK) {
    const int e   = expert_of[t];
    const int pos = offs[e] + atomicAdd(&fill[e], 1);
    perm[pos] = t;
  }
}

// f32 -> f16: shared copy (token order) + score-scaled gathered copy (grouped order)
__global__ __launch_bounds__(256) void convert_kernel(
    const float* __restrict__ x, const int* __restrict__ perm,
    const float* __restrict__ score, _Float16* __restrict__ xb_sh,
    _Float16* __restrict__ xb_rt) {
  const int row = blockIdx.x;                 // 0..T-1
  const float* xs = x + (size_t)row * HDIM;
  _Float16* ds = xb_sh + (size_t)row * HDIM;
  for (int h = threadIdx.x; h < HDIM; h += 256) ds[h] = (_Float16)xs[h];
  const int   t  = perm[row];
  const float sc = score[t];
  const float* xr = x + (size_t)t * HDIM;
  _Float16* dr = xb_rt + (size_t)row * HDIM;
  for (int h = threadIdx.x; h < HDIM; h += 256) dr[h] = (_Float16)(xr[h] * sc);
}

// ---------------- fused gate+up GEMM + SwiGLU (double-buffered pipeline) ---------
// grid: x = IDIM/BN, y = 8 m-tiles, z = 0..7 experts (grouped rows) or 8 = shared
__global__ __launch_bounds__(256) void gemm_gateup_kernel(
    const _Float16* __restrict__ xb_rt, const _Float16* __restrict__ xb_sh,
    const float* __restrict__ gate_up_proj,     // [E,H,2I]
    const float* __restrict__ shared_gate_w,    // [H,I]
    const float* __restrict__ shared_up_w,      // [H,I]
    _Float16* __restrict__ act_rt, _Float16* __restrict__ act_sh,
    const int* __restrict__ counts, const int* __restrict__ offs) {
  const int z  = blockIdx.z;
  const int mb = blockIdx.y * BM;
  const int nb = blockIdx.x * BN;

  const _Float16* A;  _Float16* act;
  const float *Bg, *Bu;
  int ldb, Mv;
  if (z < NEXP) {
    const int cnt = counts[z];
    if (mb >= cnt) return;                     // block-uniform
    Mv = cnt - mb; if (Mv > BM) Mv = BM;
    const int rb = offs[z] + mb;
    A   = xb_rt  + (size_t)rb * HDIM;
    act = act_rt + (size_t)rb * IDIM;
    Bg  = gate_up_proj + (size_t)z * HDIM * (2 * IDIM);
    Bu  = Bg + IDIM;
    ldb = 2 * IDIM;
  } else {
    Mv  = BM;
    A   = xb_sh  + (size_t)mb * HDIM;
    act = act_sh + (size_t)mb * IDIM;
    Bg  = shared_gate_w;
    Bu  = shared_up_w;
    ldb = IDIM;
  }

  __shared__ _Float16 lA [2 * ASZ];   // 32 KB
  __shared__ _Float16 lBg[2 * BSZ];   // 16 KB
  __shared__ _Float16 lBu[2 * BSZ];   // 16 KB

  const int lane = threadIdx.x & 31;
  const int wave = threadIdx.x >> 5;
  const int m0   = wave * 16;

  // A-stage mapping: thread covers row ar, halves [ah, ah+32)
  const int ar = threadIdx.x & 127;
  const int ah = (threadIdx.x >> 7) << 5;     // 0 or 32
  // B-stage mapping
  const int bk = threadIdx.x >> 2;            // 0..63
  const int bc = (threadIdx.x & 3) << 4;      // 0,16,32,48

  // zero-fill padded rows ONCE in BOTH buffers; async loads only touch rows < Mv
  if (Mv < BM && ar >= Mv) {
    const uint4 zz = make_uint4(0, 0, 0, 0);
#pragma unroll
    for (int b = 0; b < 2; ++b) {
      uint4* d = (uint4*)(lA + b * ASZ + ar * TK + ah);
      d[0] = zz; d[1] = zz; d[2] = zz; d[3] = zz;
    }
  }

  auto issueA = [&](int k0, int buf) {
    if (ar < Mv) {
      const _Float16* src = A + (size_t)ar * HDIM + k0 + ah;
      _Float16*       dst = lA + buf * ASZ + ar * TK + ah;
      async_copy_b128(src,      dst);
      async_copy_b128(src + 8,  dst + 8);
      async_copy_b128(src + 16, dst + 16);
      async_copy_b128(src + 24, dst + 24);
    }
  };

  float4 gv0, gv1, gv2, gv3, uv0, uv1, uv2, uv3;
  auto loadB = [&](int k0) {
    const float* gs = Bg + (size_t)(k0 + bk) * ldb + nb + bc;
    const float* us = Bu + (size_t)(k0 + bk) * ldb + nb + bc;
    gv0 = *(const float4*)(gs);      gv1 = *(const float4*)(gs + 4);
    gv2 = *(const float4*)(gs + 8);  gv3 = *(const float4*)(gs + 12);
    uv0 = *(const float4*)(us);      uv1 = *(const float4*)(us + 4);
    uv2 = *(const float4*)(us + 8);  uv3 = *(const float4*)(us + 12);
  };
  auto storeB = [&](int buf) {
    store_b_tile(lBg + buf * BSZ + bc * TK + bk, gv0, gv1, gv2, gv3);
    store_b_tile(lBu + buf * BSZ + bc * TK + bk, uv0, uv1, uv2, uv3);
  };

  v8f cg[4], cu[4];
#pragma unroll
  for (int i = 0; i < 4; ++i)
#pragma unroll
    for (int j = 0; j < 8; ++j) { cg[i][j] = 0.f; cu[i][j] = 0.f; }

  // ---- prologue: fill stage 0 ----
  issueA(0, 0);
  loadB(0);
  storeB(0);
  wait_async0();
  __syncthreads();

  for (int k0 = 0; k0 < HDIM; k0 += TK) {
    const int  cur  = (k0 >> 6) & 1;
    const int  nk   = k0 + TK;
    const bool more = nk < HDIM;

    if (more) {
      issueA(nk, cur ^ 1);       // DMA next A tile while we compute
      loadB(nk);                 // next weight tile loads in flight during WMMAs
      if (nk + TK < HDIM)        // warm L2 one more stage ahead
        __builtin_prefetch(Bg + (size_t)(nk + TK + bk) * ldb + nb + bc, 0, 1);
    }

    const _Float16* cA  = lA  + cur * ASZ;
    const _Float16* cBg = lBg + cur * BSZ;
    const _Float16* cBu = lBu + cur * BSZ;
#pragma unroll
    for (int ks = 0; ks < 2; ++ks) {
      const v16h a = ld_frag_a(cA, m0, ks, lane);
#pragma unroll
      for (int nt = 0; nt < 4; ++nt) {
        cg[nt] = wmma16(a, ld_frag_b(cBg, nt * 16, ks, lane), cg[nt]);
        cu[nt] = wmma16(a, ld_frag_b(cBu, nt * 16, ks, lane), cu[nt]);
      }
    }

    if (more) storeB(cur ^ 1);   // convert+store next weights (reads done above)
    wait_async0();
    __syncthreads();
  }

  // SwiGLU epilogue: act = silu(gate) * up  (f16)
#pragma unroll
  for (int nt = 0; nt < 4; ++nt) {
#pragma unroll
    for (int j = 0; j < 8; ++j) {
      const int mloc = m0 + j + ((lane & 16) ? 8 : 0);
      if (mloc < Mv) {
        const float g = cg[nt][j];
        const float u = cu[nt][j];
        const float s = g / (1.f + __expf(-g));
        act[(size_t)mloc * IDIM + nb + nt * 16 + (lane & 15)] = (_Float16)(s * u);
      }
    }
  }
}

// ---------------- down GEMM (shared: write; routed: scatter-add) ----------------
// grid: x = HDIM/BN, y = 8 m-tiles, z = experts (routed) or 1 (shared)
__global__ __launch_bounds__(256) void gemm_down_kernel(
    const _Float16* __restrict__ actbase, const float* __restrict__ Bbase,
    float* __restrict__ out, const int* __restrict__ counts,
    const int* __restrict__ offs, const int* __restrict__ perm, int routed) {
  const int nb = blockIdx.x * BN;
  const int mb = blockIdx.y * BM;

  const _Float16* A; const float* B; const int* rowmap = nullptr;
  int Mv, rowbase = 0;
  if (routed) {
    const int e   = blockIdx.z;
    const int cnt = counts[e];
    if (mb >= cnt) return;
    Mv = cnt - mb; if (Mv > BM) Mv = BM;
    const int rb = offs[e] + mb;
    A = actbase + (size_t)rb * IDIM;
    B = Bbase + (size_t)e * IDIM * HDIM;
    rowmap = perm + rb;
  } else {
    Mv = BM;
    A = actbase + (size_t)mb * IDIM;
    B = Bbase;
    rowbase = mb;
  }

  __shared__ _Float16 lA[2 * ASZ];   // 32 KB
  __shared__ _Float16 lB[2 * BSZ];   // 16 KB

  const int lane = threadIdx.x & 31;
  const int wave = threadIdx.x >> 5;
  const int m0   = wave * 16;

  const int ar = threadIdx.x & 127;
  const int ah = (threadIdx.x >> 7) << 5;
  const int bk = threadIdx.x >> 2;
  const int bc = (threadIdx.x & 3) << 4;

  if (Mv < BM && ar >= Mv) {
    const uint4 zz = make_uint4(0, 0, 0, 0);
#pragma unroll
    for (int b = 0; b < 2; ++b) {
      uint4* d = (uint4*)(lA + b * ASZ + ar * TK + ah);
      d[0] = zz; d[1] = zz; d[2] = zz; d[3] = zz;
    }
  }

  auto issueA = [&](int k0, int buf) {
    if (ar < Mv) {
      const _Float16* src = A + (size_t)ar * IDIM + k0 + ah;
      _Float16*       dst = lA + buf * ASZ + ar * TK + ah;
      async_copy_b128(src,      dst);
      async_copy_b128(src + 8,  dst + 8);
      async_copy_b128(src + 16, dst + 16);
      async_copy_b128(src + 24, dst + 24);
    }
  };

  float4 bv0, bv1, bv2, bv3;
  auto loadB = [&](int k0) {
    const float* bs = B + (size_t)(k0 + bk) * HDIM + nb + bc;
    bv0 = *(const float4*)(bs);      bv1 = *(const float4*)(bs + 4);
    bv2 = *(const float4*)(bs + 8);  bv3 = *(const float4*)(bs + 12);
  };
  auto storeB = [&](int buf) {
    store_b_tile(lB + buf * BSZ + bc * TK + bk, bv0, bv1, bv2, bv3);
  };

  v8f c[4];
#pragma unroll
  for (int i = 0; i < 4; ++i)
#pragma unroll
    for (int j = 0; j < 8; ++j) c[i][j] = 0.f;

  issueA(0, 0);
  loadB(0);
  storeB(0);
  wait_async0();
  __syncthreads();

  for (int k0 = 0; k0 < IDIM; k0 += TK) {
    const int  cur  = (k0 >> 6) & 1;
    const int  nk   = k0 + TK;
    const bool more = nk < IDIM;

    if (more) {
      issueA(nk, cur ^ 1);
      loadB(nk);
      if (nk + TK < IDIM)
        __builtin_prefetch(B + (size_t)(nk + TK + bk) * HDIM + nb + bc, 0, 1);
    }

    const _Float16* cA = lA + cur * ASZ;
    const _Float16* cB = lB + cur * BSZ;
#pragma unroll
    for (int ks = 0; ks < 2; ++ks) {
      const v16h a = ld_frag_a(cA, m0, ks, lane);
#pragma unroll
      for (int nt = 0; nt < 4; ++nt)
        c[nt] = wmma16(a, ld_frag_b(cB, nt * 16, ks, lane), c[nt]);
    }

    if (more) storeB(cur ^ 1);
    wait_async0();
    __syncthreads();
  }

#pragma unroll
  for (int nt = 0; nt < 4; ++nt) {
#pragma unroll
    for (int j = 0; j < 8; ++j) {
      const int mloc = m0 + j + ((lane & 16) ? 8 : 0);
      if (mloc < Mv) {
        const int row = routed ? rowmap[mloc] : (rowbase + mloc);
        const size_t idx = (size_t)row * HDIM + nb + nt * 16 + (lane & 15);
        const float v = c[nt][j];
        out[idx] = routed ? (out[idx] + v) : v;   // disjoint token rows -> race-free
      }
    }
  }
}

// ---------------- host launcher ----------------
extern "C" void kernel_launch(void* const* d_in, const int* in_sizes, int n_in,
                              void* d_out, int out_size, void* d_ws, size_t ws_size,
                              hipStream_t stream) {
  const float* x       = (const float*)d_in[0];   // [T,H]
  const float* rw      = (const float*)d_in[1];   // [H,E]
  const float* gate_up = (const float*)d_in[2];   // [E,H,2I]
  const float* down    = (const float*)d_in[3];   // [E,I,H]
  const float* sg      = (const float*)d_in[4];   // [H,I]
  const float* su      = (const float*)d_in[5];   // [H,I]
  const float* sd      = (const float*)d_in[6];   // [I,H]

  float* out    = (float*)d_out;                  // [T,H]
  float* scores = out + (size_t)T_TOK * HDIM;     // [E,T]

  // workspace carve-up (~24 MB)
  char* ws = (char*)d_ws;
  size_t p = 0;
  int*   expert_of = (int*)  (ws + p); p += (size_t)T_TOK * 4;
  float* score     = (float*)(ws + p); p += (size_t)T_TOK * 4;
  int*   counts    = (int*)  (ws + p); p += 256;
  int*   offs      = (int*)  (ws + p); p += 256;
  int*   fill      = (int*)  (ws + p); p += 256;
  int*   perm      = (int*)  (ws + p); p += (size_t)T_TOK * 4;
  p = (p + 255) & ~(size_t)255;
  _Float16* xb_sh  = (_Float16*)(ws + p); p += (size_t)T_TOK * HDIM * 2;
  _Float16* xb_rt  = (_Float16*)(ws + p); p += (size_t)T_TOK * HDIM * 2;
  _Float16* act_sh = (_Float16*)(ws + p); p += (size_t)T_TOK * IDIM * 2;
  _Float16* act_rt = (_Float16*)(ws + p); p += (size_t)T_TOK * IDIM * 2;
  (void)ws_size; (void)n_in; (void)in_sizes; (void)out_size;

  moe_init_kernel<<<1, 32, 0, stream>>>(counts, fill);
  router_kernel<<<T_TOK, 256, 0, stream>>>(x, rw, scores, expert_of, score, counts);
  offsets_kernel<<<1, 32, 0, stream>>>(counts, offs);
  scatter_kernel<<<(T_TOK + 255) / 256, 256, 0, stream>>>(expert_of, offs, fill, perm);
  convert_kernel<<<T_TOK, 256, 0, stream>>>(x, perm, score, xb_sh, xb_rt);

  // gate+up+SwiGLU: z = 8 experts + 1 shared pass
  gemm_gateup_kernel<<<dim3(IDIM / BN, T_TOK / BM, NEXP + 1), 256, 0, stream>>>(
      xb_rt, xb_sh, gate_up, sg, su, act_rt, act_sh, counts, offs);

  // down: shared writes out, then routed experts scatter-add
  gemm_down_kernel<<<dim3(HDIM / BN, T_TOK / BM, 1), 256, 0, stream>>>(
      act_sh, sd, out, counts, offs, perm, 0);
  gemm_down_kernel<<<dim3(HDIM / BN, T_TOK / BM, NEXP), 256, 0, stream>>>(
      act_rt, down, out, counts, offs, perm, 1);
}
